// Ses2Seq_27719718928984
// MI455X (gfx1250) — matrix operand order repaired and checked by reading, not compiled, not run on hardware
//
#include <hip/hip_runtime.h>

// ---------------------------------------------------------------------------
// MI455X plan (compile-only reasoning):
//  * 256 sequential steps of C(16x9291) = Theta(16x9291) @ A^T + x_t @ Bm^T.
//  * HBM-bound: A must be streamed each step. Keep it at 4 B/elem like f32,
//    but as TWO bf16 planes (hi + round-residual lo) so the math runs on
//    v_wmma_f32_16x16x32_bf16 with ~f32 accuracy via 3 product terms:
//        Th_hi*A_hi + Th_lo*A_hi + Th_hi*A_lo
//    -> ~186 TF effective ceiling at 23.3 TB/s; A_hi (173 MB) can even sit in
//       the 192 MB L2 across steps.
//  * M dimension of the 16x16x32 WMMA == batch (16): perfect fit, one wave
//    per 16-column output tile, 291 K-chunks, f32 accumulation in VGPRs.
//  * Theta_next is written as f32 (for the final MLP) plus a fresh bf16 hi/lo
//    split (A-operand of the next step). Ping-pong buffers in d_ws.
//  * Tiny MLP tail: one block per batch row, Theta row staged in LDS (37 KB
//    of the 320 KB/WGP), scalar f32 FMAs (negligible cost).
// ---------------------------------------------------------------------------

typedef __bf16 bf16_t;
typedef __attribute__((ext_vector_type(8)))  __bf16 v8bf;
typedef __attribute__((ext_vector_type(16))) __bf16 v16bf;
typedef __attribute__((ext_vector_type(8)))  float  v8f;

#define L_DIM   9291
#define KP      9312          // K padded to multiple of 32 (291 chunks)
#define NP      9296          // N padded to multiple of 16 (581 tiles)
#define NTILES  581
#define BATCH   16
#define TSTEPS  256

// ---------------- A -> (A_hi, A_lo) bf16 split, zero-padded -----------------
__global__ __launch_bounds__(256) void convert_A_kernel(
    const float* __restrict__ A, bf16_t* __restrict__ A_hi, bf16_t* __restrict__ A_lo)
{
    size_t idx = (size_t)blockIdx.x * 256 + threadIdx.x;
    if (idx >= (size_t)NP * KP) return;
    int n = (int)(idx / KP);
    int k = (int)(idx % KP);
    float v = (n < L_DIM && k < L_DIM) ? A[(size_t)n * L_DIM + k] : 0.0f;
    bf16_t hi = (bf16_t)v;
    bf16_t lo = (bf16_t)(v - (float)hi);
    A_hi[idx] = hi;
    A_lo[idx] = lo;
}

// --------- Theta0 = broadcast(theta) in ping buffers; pong zeroed -----------
__global__ __launch_bounds__(256) void init_theta_kernel(
    const float* __restrict__ theta,
    bf16_t* __restrict__ ping_hi, bf16_t* __restrict__ ping_lo,
    bf16_t* __restrict__ pong_hi, bf16_t* __restrict__ pong_lo)
{
    int idx = blockIdx.x * 256 + threadIdx.x;
    if (idx >= BATCH * KP) return;
    int k = idx % KP;
    float v = (k < L_DIM) ? theta[k] : 0.0f;
    bf16_t hi = (bf16_t)v;
    bf16_t lo = (bf16_t)(v - (float)hi);
    ping_hi[idx] = hi;
    ping_lo[idx] = lo;
    pong_hi[idx] = (bf16_t)0.0f;   // keep pad columns exactly zero forever
    pong_lo[idx] = (bf16_t)0.0f;
}

// ------------- one recurrence step: Theta_out = Theta_in @ A^T + x_t Bm^T ---
__global__ __launch_bounds__(128) void step_kernel(
    const bf16_t* __restrict__ th_hi, const bf16_t* __restrict__ th_lo,
    const bf16_t* __restrict__ A_hi,  const bf16_t* __restrict__ A_lo,
    const float*  __restrict__ xs,    const float*  __restrict__ Bm, int t,
    bf16_t* __restrict__ out_hi, bf16_t* __restrict__ out_lo,
    float*  __restrict__ out_f32)
{
    const int lane = threadIdx.x & 31;
    const int wave = threadIdx.x >> 5;
    const int tile = blockIdx.x * 4 + wave;
    if (tile >= NTILES) return;              // wave-uniform: EXEC stays all-1s

    const int nb   = tile * 16;
    const int ksel = lane >> 4;              // half-wave select
    const int ml   = lane & 15;              // M row (A-op) / N col (B-op)

    const bf16_t* __restrict__ thh = th_hi + ml * KP;
    const bf16_t* __restrict__ thl = th_lo + ml * KP;
    const bf16_t* __restrict__ arh = A_hi + (size_t)(nb + ml) * KP;
    const bf16_t* __restrict__ arl = A_lo + (size_t)(nb + ml) * KP;

    v8f acc = {};
    for (int kk = 0; kk < KP; kk += 32) {
        // A-operand (Theta 16x32): per ISA layout, lane needs K runs
        // [kk + ksel*8 .. +7] and [kk + 16 + ksel*8 .. +7]
        v8bf ah0 = *(const v8bf*)(thh + kk + ksel * 8);
        v8bf ah1 = *(const v8bf*)(thh + kk + 16 + ksel * 8);
        v8bf al0 = *(const v8bf*)(thl + kk + ksel * 8);
        v8bf al1 = *(const v8bf*)(thl + kk + 16 + ksel * 8);
        v16bf a_hi = __builtin_shufflevector(ah0, ah1, 0,1,2,3,4,5,6,7,8,9,10,11,12,13,14,15);
        v16bf a_lo = __builtin_shufflevector(al0, al1, 0,1,2,3,4,5,6,7,8,9,10,11,12,13,14,15);

        // B-operand (A^T 32x16): lane = column n, K runs [kk + ksel*8 .. +7] and [kk + 16 + ksel*8 .. +7] from row n
        v8bf bh0 = *(const v8bf*)(arh + kk + ksel * 8);
        v8bf bh1 = *(const v8bf*)(arh + kk + 16 + ksel * 8);
        v8bf bl0 = *(const v8bf*)(arl + kk + ksel * 8);
        v8bf bl1 = *(const v8bf*)(arl + kk + 16 + ksel * 8);
        v16bf b_hi = __builtin_shufflevector(bh0, bh1, 0,1,2,3,4,5,6,7,8,9,10,11,12,13,14,15);
        v16bf b_lo = __builtin_shufflevector(bl0, bl1, 0,1,2,3,4,5,6,7,8,9,10,11,12,13,14,15);

        // speculative prefetch of next A rows (global_prefetch_b8)
        __builtin_prefetch(arh + kk + 32, 0, 1);
        __builtin_prefetch(arl + kk + 32, 0, 1);

        // 3-term hi/lo split product ~= f32 GEMM on the bf16 WMMA pipe
        acc = __builtin_amdgcn_wmma_f32_16x16x32_bf16(false, a_hi, false, b_hi, (short)0, acc, false, false);
        acc = __builtin_amdgcn_wmma_f32_16x16x32_bf16(false, a_lo, false, b_hi, (short)0, acc, false, false);
        acc = __builtin_amdgcn_wmma_f32_16x16x32_bf16(false, a_hi, false, b_lo, (short)0, acc, false, false);
    }

    // epilogue: rank-1 bias x_t * Bm^T, write f32 + fresh bf16 hi/lo split
    const int n   = nb + ml;                     // C layout: N = lane & 15
    const float bmn = (n < L_DIM) ? Bm[n] : 0.0f;
#pragma unroll
    for (int r = 0; r < 8; ++r) {
        const int m = r + ksel * 8;              // C layout: M = r + 8*(lane>>4)
        float v = acc[r] + xs[m * TSTEPS + t] * bmn;
        out_f32[m * KP + n] = v;
        bf16_t hi = (bf16_t)v;
        out_hi[m * KP + n] = hi;
        out_lo[m * KP + n] = (bf16_t)(v - (float)hi);
    }
}

// ------------------------- MLP tail (two heads) -----------------------------
// block = (b); 256 threads = timesteps; Theta row staged in LDS.
__global__ __launch_bounds__(256) void mlp_kernel(
    const float* __restrict__ Th, float* __restrict__ out)
{
    __shared__ float s[L_DIM];
    const int b = blockIdx.x;
    const float* row = Th + (size_t)b * KP;
    for (int i = threadIdx.x; i < L_DIM; i += 256) s[i] = row[i];
    __syncthreads();

    const int t = threadIdx.x;
    const float ts = (float)t / (float)(TSTEPS - 1);

    float h1[64];
    // ----- recon head: offsets W1r=0 b1r=64 W2r=128 b2r=4224 W3r=4288 b3r=4352
#pragma unroll
    for (int o = 0; o < 64; ++o) {
        float a = s[o] * ts + s[64 + o];
        h1[o] = a > 0.f ? a : 0.f;
    }
    float outr = s[4352];
    for (int o = 0; o < 64; ++o) {
        float a = s[4224 + o];
#pragma unroll
        for (int i = 0; i < 64; ++i) a += s[128 + o * 64 + i] * h1[i];
        float h2 = a > 0.f ? a : 0.f;
        outr += s[4288 + o] * h2;
    }
    out[b * TSTEPS + t] = outr;

    // ----- class head: W1c=4353 b1c=4417 W2c=4481 b2c=8577 W3c=8641 b3c=9281
#pragma unroll
    for (int o = 0; o < 64; ++o) {
        float a = s[4353 + o] * ts + s[4417 + o];
        h1[o] = a > 0.f ? a : 0.f;
    }
    float outc[10];
#pragma unroll
    for (int c = 0; c < 10; ++c) outc[c] = s[9281 + c];
    for (int o = 0; o < 64; ++o) {
        float a = s[8577 + o];
#pragma unroll
        for (int i = 0; i < 64; ++i) a += s[4481 + o * 64 + i] * h1[i];
        float h2 = a > 0.f ? a : 0.f;
#pragma unroll
        for (int c = 0; c < 10; ++c) outc[c] += s[8641 + c * 64 + o] * h2;
    }
    float* oc = out + BATCH * TSTEPS;
#pragma unroll
    for (int c = 0; c < 10; ++c) oc[(b * TSTEPS + t) * 10 + c] = outc[c];
}

// ---------------------------------------------------------------------------
extern "C" void kernel_launch(void* const* d_in, const int* in_sizes, int n_in,
                              void* d_out, int out_size, void* d_ws, size_t ws_size,
                              hipStream_t stream) {
    const float* xs    = (const float*)d_in[0];   // (16, 256, 1)
    const float* A     = (const float*)d_in[1];   // (9291, 9291)
    const float* Bm    = (const float*)d_in[2];   // (9291, 1)
    const float* theta = (const float*)d_in[3];   // (9291,)
    float* out = (float*)d_out;                   // 4096 + 40960 f32

    // workspace carve (all sizes are multiples of 256 B)
    char* ws = (char*)d_ws;
    const size_t A_bytes  = (size_t)NP * KP * sizeof(bf16_t);   // 173,128,704
    const size_t th_bytes = (size_t)BATCH * KP * sizeof(bf16_t); //    297,984
    bf16_t* A_hi  = (bf16_t*)(ws);
    bf16_t* A_lo  = (bf16_t*)(ws + A_bytes);
    bf16_t* th_hi[2] = { (bf16_t*)(ws + 2 * A_bytes),
                         (bf16_t*)(ws + 2 * A_bytes + 2 * th_bytes) };
    bf16_t* th_lo[2] = { (bf16_t*)(ws + 2 * A_bytes + th_bytes),
                         (bf16_t*)(ws + 2 * A_bytes + 3 * th_bytes) };
    float*  th_f32   = (float*) (ws + 2 * A_bytes + 4 * th_bytes);

    // 1) split A into bf16 hi/lo planes (zero-padded)
    {
        size_t n = (size_t)NP * KP;
        int grid = (int)((n + 255) / 256);
        convert_A_kernel<<<grid, 256, 0, stream>>>(A, A_hi, A_lo);
    }
    // 2) Theta0 broadcast + zero both ping/pong (pad columns stay 0)
    {
        int grid = (BATCH * KP + 255) / 256;
        init_theta_kernel<<<grid, 256, 0, stream>>>(theta, th_hi[0], th_lo[0],
                                                    th_hi[1], th_lo[1]);
    }
    // 3) 256 sequential recurrence steps (ping-pong)
    for (int t = 0; t < TSTEPS; ++t) {
        int cur = t & 1, nxt = cur ^ 1;
        step_kernel<<<(NTILES + 3) / 4, 128, 0, stream>>>(
            th_hi[cur], th_lo[cur], A_hi, A_lo, xs, Bm, t,
            th_hi[nxt], th_lo[nxt], th_f32);
    }
    // 4) MLP heads from Theta_final (f32)
    mlp_kernel<<<BATCH, 256, 0, stream>>>(th_f32, out);
}
